// Attention_32134945308847
// MI455X (gfx1250) — matrix operand order repaired and checked by reading, not compiled, not run on hardware
//
#include <hip/hip_runtime.h>
#include <hip/hip_bf16.h>
#include <stdint.h>

// ---------------------------------------------------------------------------
// Types for CDNA5 WMMA (gfx1250, wave32)
// ---------------------------------------------------------------------------
typedef __attribute__((ext_vector_type(16))) __bf16 v16bf;
typedef __attribute__((ext_vector_type(8)))  __bf16 v8bf;
typedef __attribute__((ext_vector_type(8)))  float  v8f;

// A-fragment (16x32, MxK): row = lane&15, K runs [kb..kb+7] and [kb+16..kb+23],
// kb = 8*(lane>>4).  Source is row-major [M, ld].
__device__ inline v16bf load_fragA(const __bf16* __restrict__ base, int ld,
                                   int row, int k0, int lane) {
    int kb = (lane >> 4) << 3;
    const __bf16* p = base + (size_t)row * ld + k0 + kb;
    v8bf lo = *(const v8bf*)p;
    v8bf hi = *(const v8bf*)(p + 16);
    return __builtin_shufflevector(lo, hi, 0, 1, 2, 3, 4, 5, 6, 7,
                                           8, 9, 10, 11, 12, 13, 14, 15);
}

// B-fragment (32x16, KxN) sourced from row-major [N, ld] storage (i.e. B^T):
// col = lane&15, K runs [kb..kb+7] and [kb+16..kb+23], kb = 8*(lane>>4).
__device__ inline v16bf load_fragB(const __bf16* __restrict__ base, int ld,
                                   int ncol, int k0, int lane) {
    int kb = (lane >> 4) << 3;
    const __bf16* p = base + (size_t)ncol * ld + k0 + kb;
    v8bf lo = *(const v8bf*)p;
    v8bf hi = *(const v8bf*)(p + 16);
    return __builtin_shufflevector(lo, hi, 0, 1, 2, 3, 4, 5, 6, 7,
                                           8, 9, 10, 11, 12, 13, 14, 15);
}

__device__ inline v8f wmma_bf16(v16bf a, v16bf b, v8f c) {
    return __builtin_amdgcn_wmma_f32_16x16x32_bf16(false, a, false, b,
                                                   (short)0, c, false, false);
}

// Reductions across each 16-lane half of a wave32 (C-fragment row groups).
__device__ inline float rowmax16(float v) {
    #pragma unroll
    for (int m = 1; m < 16; m <<= 1) v = fmaxf(v, __shfl_xor(v, m, 32));
    return v;
}
__device__ inline float rowsum16(float v) {
    #pragma unroll
    for (int m = 1; m < 16; m <<= 1) v += __shfl_xor(v, m, 32);
    return v;
}

// CDNA5 async global->LDS copy (16B per lane), tracked by ASYNCcnt.
__device__ inline void cp_async_b128(void* lds_ptr, const void* gptr) {
    unsigned loff = (unsigned)(uintptr_t)lds_ptr;   // low 32b of generic = LDS offset
    asm volatile("global_load_async_to_lds_b128 %0, %1, off"
                 :: "v"(loff), "v"(gptr)
                 : "memory");
}
__device__ inline void wait_async0() {
    asm volatile("s_wait_asynccnt 0x0" ::: "memory");
}

// ---------------------------------------------------------------------------
// Elementwise conversion kernels
// ---------------------------------------------------------------------------
__global__ __launch_bounds__(256) void f32_to_bf16_kernel(
    const float* __restrict__ in, __bf16* __restrict__ out, size_t n) {
    size_t i = (size_t)blockIdx.x * blockDim.x + threadIdx.x;
    if (i < n) out[i] = (__bf16)in[i];
}

// in: row-major [R, C] f32 -> out: row-major [C, R] bf16
__global__ __launch_bounds__(256) void transpose_to_bf16_kernel(
    const float* __restrict__ in, __bf16* __restrict__ out, int R, int C) {
    size_t i = (size_t)blockIdx.x * blockDim.x + threadIdx.x;
    if (i < (size_t)R * C) {
        int r = (int)(i / C);
        int c = (int)(i % C);
        out[(size_t)c * R + r] = (__bf16)in[i];
    }
}

// ---------------------------------------------------------------------------
// Tiled GEMM: C[M,N] = A[M,K](bf16) @ B[K,N] + bias, B stored as Bt[N,K] bf16.
// 256 threads (8 waves). Block tile 128x128, K-step 32, double-buffered LDS
// filled with async global->LDS b128 copies. Each wave computes 32x64
// (2 A-frags x 4 B-frags = 8 WMMAs per K-step).
// ---------------------------------------------------------------------------
#define GT_M 128
#define GT_N 128
#define GT_K 32

__global__ __launch_bounds__(256) void gemm_bf16_tiled_kernel(
    const __bf16* __restrict__ A, const __bf16* __restrict__ Bt,
    const float* __restrict__ bias, float* __restrict__ C,
    int M, int N, int K) {
    __shared__ __align__(16) __bf16 sA[2][GT_M * GT_K];
    __shared__ __align__(16) __bf16 sB[2][GT_N * GT_K];

    const int t = threadIdx.x;
    const int lane = t & 31;
    const int wave = t >> 5;             // 0..7
    const int mw = wave & 3;             // 4 M sub-tiles of 32
    const int nw = wave >> 2;            // 2 N sub-tiles of 64
    const int m_off = mw * 32;
    const int n_off = nw * 64;
    const int n0 = blockIdx.x * GT_N;
    const int m0 = blockIdx.y * GT_M;
    const int colk = lane & 15;
    const int kb8  = (lane >> 4) << 3;
    const int kb16 = (lane >> 4) << 4;

    // Stage one 128x32 A tile + 128x32 B tile: 512 16B-chunks each,
    // 2 chunks per thread per tile.
    auto stage = [&](int buf, int k0) {
        #pragma unroll
        for (int c2 = 0; c2 < 2; ++c2) {
            int c = t + c2 * 256;        // 0..511
            int row = c >> 2;            // 0..127
            int koff = (c & 3) * 8;      // elements
            cp_async_b128(&sA[buf][c * 8], A  + (size_t)(m0 + row) * K + k0 + koff);
            cp_async_b128(&sB[buf][c * 8], Bt + (size_t)(n0 + row) * K + k0 + koff);
        }
    };

    v8f acc[2][4];
    #pragma unroll
    for (int i = 0; i < 2; ++i)
        #pragma unroll
        for (int j = 0; j < 4; ++j)
            #pragma unroll
            for (int r = 0; r < 8; ++r) acc[i][j][r] = 0.0f;

    stage(0, 0);
    wait_async0();
    __syncthreads();

    const int nk = K / GT_K;
    for (int kk = 0; kk < nk; ++kk) {
        int cur = kk & 1;
        if (kk + 1 < nk) stage(cur ^ 1, (kk + 1) * GT_K);

        // fragments from LDS
        v16bf af[2], bfv[4];
        #pragma unroll
        for (int i = 0; i < 2; ++i) {
            const __bf16* p = &sA[cur][(m_off + i * 16 + colk) * GT_K + kb8];
            v8bf lo = *(const v8bf*)p;
            v8bf hi = *(const v8bf*)(p + 16);
            af[i] = __builtin_shufflevector(lo, hi, 0, 1, 2, 3, 4, 5, 6, 7,
                                                    8, 9, 10, 11, 12, 13, 14, 15);
        }
        #pragma unroll
        for (int j = 0; j < 4; ++j) {
            const __bf16* p = &sB[cur][(n_off + j * 16 + colk) * GT_K + kb8];
            v8bf lo = *(const v8bf*)p;
            v8bf hi = *(const v8bf*)(p + 16);
            bfv[j] = __builtin_shufflevector(lo, hi, 0, 1, 2, 3, 4, 5, 6, 7,
                                                     8, 9, 10, 11, 12, 13, 14, 15);
        }
        #pragma unroll
        for (int i = 0; i < 2; ++i)
            #pragma unroll
            for (int j = 0; j < 4; ++j)
                acc[i][j] = wmma_bf16(af[i], bfv[j], acc[i][j]);

        wait_async0();
        __syncthreads();
    }

    const int rowoff = (lane >> 4) << 3;
    #pragma unroll
    for (int i = 0; i < 2; ++i) {
        #pragma unroll
        for (int j = 0; j < 4; ++j) {
            int n = n0 + n_off + j * 16 + colk;
            float bv = bias[n];
            #pragma unroll
            for (int r = 0; r < 8; ++r) {
                int mrow = m0 + m_off + i * 16 + r + rowoff;
                C[(size_t)mrow * N + n] = acc[i][j][r] + bv;
            }
        }
    }
}

// ---------------------------------------------------------------------------
// RoPE + head split.  qkv: f32 [B*S, 3*D], per row head h at col h*384:
// q = [0,128), k = [128,256), v = [256,384).
// NOTE: reference swaps cos/sin:  cos := sin(emb), sin := cos(emb).
// Q is pre-scaled by 1/sqrt(hd).
// Outputs: Qh,Kh bf16 [B*H, S, hd], Vt bf16 [B*H, hd, S] (transposed).
// ---------------------------------------------------------------------------
__global__ __launch_bounds__(256) void rope_split_kernel(
    const float* __restrict__ qkv, __bf16* __restrict__ Qh,
    __bf16* __restrict__ Kh, __bf16* __restrict__ Vt) {
    const int S = 2048, hd = 128;
    size_t t = (size_t)blockIdx.x * blockDim.x + threadIdx.x;
    int i = (int)(t & 127);
    size_t rest = t >> 7;          // (b*H + h)*S + s
    int s = (int)(rest & 2047);
    int bh = (int)(rest >> 11);    // 0..31
    int h = bh & 15;
    int b = bh >> 4;

    size_t rowbase = ((size_t)(b * S + s)) * 6144 + (size_t)h * 384;
    float q = qkv[rowbase + i];
    float k = qkv[rowbase + 128 + i];
    float v = qkv[rowbase + 256 + i];
    int f = i & 63;
    // inv_freq[f] = 10000^(-2f/hd) = exp(-ln(10000)*2f/128)
    float angle = (float)s * __expf(-(2.0f * (float)f / 128.0f) * 9.210340371976184f);
    float cosv = __sinf(angle);    // reference swap
    float sinv = __cosf(angle);    // reference swap
    float qp = (i < 64) ? -qkv[rowbase + i + 64]       : qkv[rowbase + i - 64];
    float kp = (i < 64) ? -qkv[rowbase + 128 + i + 64] : qkv[rowbase + 128 + i - 64];
    float qo = (q * cosv + qp * sinv) * 0.08838834764831845f; // 1/sqrt(128)
    float ko = k * cosv + kp * sinv;

    size_t o = ((size_t)bh * S + s) * hd + i;
    Qh[o] = (__bf16)qo;
    Kh[o] = (__bf16)ko;
    Vt[((size_t)bh * hd + i) * S + s] = (__bf16)v;
}

// ---------------------------------------------------------------------------
// Causal flash attention: one wave per (b,h, 16-query tile).
// Online softmax; 32 keys per step (2 score tiles); P transposed via LDS.
// ---------------------------------------------------------------------------
__global__ __launch_bounds__(32) void flash_kernel(
    const __bf16* __restrict__ Qh, const __bf16* __restrict__ Kh,
    const __bf16* __restrict__ Vt, __bf16* __restrict__ attn) {
    const int S = 2048, hd = 128, D = 2048;
    __shared__ __align__(16) __bf16 plds[16 * 32];

    int lane = threadIdx.x & 31;
    int qb = blockIdx.x;            // 0..127
    int bh = blockIdx.y;            // 0..31
    int q0 = qb * 16;
    int b = bh >> 4;
    int h = bh & 15;
    const __bf16* Qb = Qh + (size_t)bh * S * hd;
    const __bf16* Kb = Kh + (size_t)bh * S * hd;
    const __bf16* Vb = Vt + (size_t)bh * hd * S;

    int colk = lane & 15;
    int rowoff = (lane >> 4) << 3;

    v16bf qf[4];
    #pragma unroll
    for (int c = 0; c < 4; ++c)
        qf[c] = load_fragA(Qb, hd, q0 + colk, c * 32, lane);

    v8f acc[8];
    float mrow[8], lrow[8];
    #pragma unroll
    for (int r = 0; r < 8; ++r) { mrow[r] = -3e38f; lrow[r] = 0.0f; }
    #pragma unroll
    for (int c = 0; c < 8; ++c)
        #pragma unroll
        for (int r = 0; r < 8; ++r) acc[c][r] = 0.0f;

    int npairs = qb / 2 + 1;        // covers key blocks [0, q0+16) or [0, q0+32)
    for (int p = 0; p < npairs; ++p) {
        int k0 = p * 32;
        v8f s0, s1;
        #pragma unroll
        for (int r = 0; r < 8; ++r) { s0[r] = 0.0f; s1[r] = 0.0f; }

        #pragma unroll
        for (int c = 0; c < 4; ++c) {
            v16bf kf0 = load_fragB(Kb, hd, k0 + colk, c * 32, lane);
            s0 = wmma_bf16(qf[c], kf0, s0);
            v16bf kf1 = load_fragB(Kb, hd, k0 + 16 + colk, c * 32, lane);
            s1 = wmma_bf16(qf[c], kf1, s1);
        }

        // online softmax update + stage P (bf16) into LDS for transpose
        #pragma unroll
        for (int r = 0; r < 8; ++r) {
            int qrow = q0 + r + rowoff;
            float v0 = (k0 + colk      <= qrow) ? s0[r] : -1e30f;
            float v1 = (k0 + 16 + colk <= qrow) ? s1[r] : -1e30f;
            float mx = rowmax16(fmaxf(v0, v1));
            mx = fmaxf(mx, mrow[r]);
            float al = __expf(mrow[r] - mx);
            mrow[r] = mx;
            float p0 = __expf(v0 - mx);
            float p1 = __expf(v1 - mx);
            lrow[r] = lrow[r] * al + rowsum16(p0 + p1);
            plds[(r + rowoff) * 32 + colk]      = (__bf16)p0;
            plds[(r + rowoff) * 32 + 16 + colk] = (__bf16)p1;
            #pragma unroll
            for (int c = 0; c < 8; ++c) acc[c][r] *= al;
        }
        __syncthreads();

        // P as A-fragment (16 rows x 32 keys) from LDS
        int kb8 = (lane >> 4) << 3;
        const __bf16* pp = &plds[colk * 32 + kb8];
        v8bf plo = *(const v8bf*)pp;
        v8bf phi = *(const v8bf*)(pp + 16);
        v16bf pf = __builtin_shufflevector(plo, phi, 0, 1, 2, 3, 4, 5, 6, 7,
                                                     8, 9, 10, 11, 12, 13, 14, 15);
        #pragma unroll
        for (int c = 0; c < 8; ++c) {
            v16bf vf = load_fragB(Vb, S, c * 16 + colk, k0, lane);
            acc[c] = wmma_bf16(pf, vf, acc[c]);
        }
        __syncthreads();
    }

    // epilogue: normalize and scatter back to [B*S, D] bf16
    #pragma unroll
    for (int r = 0; r < 8; ++r) {
        float inv = 1.0f / lrow[r];
        size_t row = (size_t)(b * S + q0 + r + rowoff);
        #pragma unroll
        for (int c = 0; c < 8; ++c)
            attn[row * D + h * hd + c * 16 + colk] = (__bf16)(acc[c][r] * inv);
    }
}

// ---------------------------------------------------------------------------
// Host-side pipeline
// ---------------------------------------------------------------------------
extern "C" void kernel_launch(void* const* d_in, const int* in_sizes, int n_in,
                              void* d_out, int out_size, void* d_ws, size_t ws_size,
                              hipStream_t stream) {
    const int Bb = 2, S = 2048, D = 2048, H = 16;
    const int BS = Bb * S;      // 4096
    const int N3 = 3 * D;       // 6144
    (void)in_sizes; (void)n_in; (void)out_size; (void)ws_size;

    const float* x     = (const float*)d_in[0];
    const float* w_qkv = (const float*)d_in[1];
    const float* b_qkv = (const float*)d_in[2];
    const float* w_o   = (const float*)d_in[3];
    const float* b_o   = (const float*)d_in[4];
    float* out = (float*)d_out;

    char* ws = (char*)d_ws;
    size_t off = 0;
    auto suballoc = [&](size_t bytes) -> void* {
        void* p = ws + off;
        off += (bytes + 255) & ~(size_t)255;
        return p;
    };
    __bf16* xbf   = (__bf16*)suballoc((size_t)BS * D * 2);
    __bf16* wqkvT = (__bf16*)suballoc((size_t)N3 * D * 2);
    __bf16* woT   = (__bf16*)suballoc((size_t)D * D * 2);
    float*  qkv   = (float*) suballoc((size_t)BS * N3 * 4);
    __bf16* Qh    = (__bf16*)suballoc((size_t)BS * D * 2);  // B*H*S*hd
    __bf16* Kh    = (__bf16*)suballoc((size_t)BS * D * 2);
    __bf16* Vt    = (__bf16*)suballoc((size_t)BS * D * 2);
    __bf16* attn  = (__bf16*)suballoc((size_t)BS * D * 2);

    {   size_t n = (size_t)BS * D;
        f32_to_bf16_kernel<<<dim3((unsigned)((n + 255) / 256)), dim3(256), 0, stream>>>(x, xbf, n);
    }
    {   size_t n = (size_t)D * N3;
        transpose_to_bf16_kernel<<<dim3((unsigned)((n + 255) / 256)), dim3(256), 0, stream>>>(w_qkv, wqkvT, D, N3);
    }
    {   size_t n = (size_t)D * D;
        transpose_to_bf16_kernel<<<dim3((unsigned)((n + 255) / 256)), dim3(256), 0, stream>>>(w_o, woT, D, D);
    }
    // qkv = x @ w_qkv + b_qkv
    gemm_bf16_tiled_kernel<<<dim3(N3 / GT_N, BS / GT_M), dim3(256), 0, stream>>>(
        xbf, wqkvT, b_qkv, qkv, BS, N3, D);
    // RoPE + head split (+ 1/sqrt(hd) folded into Q)
    {   size_t n = (size_t)BS * D;   // B*H*S*hd
        rope_split_kernel<<<dim3((unsigned)(n / 256)), dim3(256), 0, stream>>>(qkv, Qh, Kh, Vt);
    }
    // causal flash attention
    flash_kernel<<<dim3(S / 16, Bb * H), dim3(32), 0, stream>>>(Qh, Kh, Vt, attn);
    // out = attn @ w_o + b_o
    gemm_bf16_tiled_kernel<<<dim3(D / GT_N, BS / GT_M), dim3(256), 0, stream>>>(
        attn, woT, b_o, out, BS, D, D);
}
